// LSTMModel_15977278341144
// MI455X (gfx1250) — matrix operand order repaired and checked by reading, not compiled, not run on hardware
//
#include <hip/hip_runtime.h>

// ---------------------------------------------------------------------------
// Types for CDNA5 WMMA
// ---------------------------------------------------------------------------
typedef __attribute__((ext_vector_type(16))) __bf16 v16bf;
typedef __attribute__((ext_vector_type(8)))  float  v8f;

union Frag {
    v16bf v;
    unsigned u[8];
    uint4 q[2];
};

// fp32 -> bf16 round-to-nearest-even
__device__ inline unsigned short f2bf(float f) {
    unsigned u = __float_as_uint(f);
    unsigned r = (u + 0x7fffu + ((u >> 16) & 1u)) >> 16;
    return (unsigned short)r;
}
__device__ inline unsigned pack_bf(float a, float b) {
    return (unsigned)f2bf(a) | ((unsigned)f2bf(b) << 16);
}
__device__ inline float sigf(float x) { return 1.0f / (1.0f + __expf(-x)); }

// ---------------------------------------------------------------------------
// Hybrid cluster/global barrier.
// Emits the CDNA5 cluster split-barrier (NOP when not cluster-dispatched) and
// is made unconditionally correct by a monotonic agent-scope atomic counter.
// One wave signals per workgroup (ISA-recommended pattern).
// ---------------------------------------------------------------------------
__device__ inline void grid_sync(unsigned* cnt, unsigned& target, int nWG) {
    __syncthreads();
    __builtin_amdgcn_fence(__ATOMIC_RELEASE, "agent");
    target += (unsigned)nWG;
    if (threadIdx.x == 0)
        __hip_atomic_fetch_add(cnt, 1u, __ATOMIC_RELAXED, __HIP_MEMORY_SCOPE_AGENT);
    if ((threadIdx.x >> 5) == 0)
        asm volatile("s_barrier_signal -3" ::: "memory");
    asm volatile("s_barrier_wait -3" ::: "memory");
    if (threadIdx.x == 0) {
        while (__hip_atomic_load(cnt, __ATOMIC_RELAXED, __HIP_MEMORY_SCOPE_AGENT) < target)
            __builtin_amdgcn_s_sleep(8);
    }
    __syncthreads();
    __builtin_amdgcn_fence(__ATOMIC_ACQUIRE, "agent");
}

// ---------------------------------------------------------------------------
// Weight repack: transposed bf16 weights + zero barrier counters.
//   wt0 [1024][384] : k<128 -> W0[k][n], else U0[k-128][n]
//   wt1 [1024][512] : k<256 -> W1[k][n], else U1[k-256][n]
//   wfcT[128][256]  : Wfc[k][n]
// ---------------------------------------------------------------------------
__global__ void convert_kernel(const float* __restrict__ W0, const float* __restrict__ U0,
                               const float* __restrict__ W1, const float* __restrict__ U1,
                               const float* __restrict__ Wfc,
                               unsigned short* __restrict__ wt0,
                               unsigned short* __restrict__ wt1,
                               unsigned short* __restrict__ wfcT,
                               unsigned* __restrict__ cnt) {
    const int N0 = 1024 * 384, N1 = 1024 * 512, N2 = 128 * 256;
    int i = blockIdx.x * 256 + threadIdx.x;
    int stride = gridDim.x * 256;
    for (int e = i; e < N0 + N1 + N2; e += stride) {
        if (e < N0) {
            int n = e / 384, k = e % 384;
            float f = (k < 128) ? W0[k * 1024 + n] : U0[(k - 128) * 1024 + n];
            wt0[n * 384 + k] = f2bf(f);
        } else if (e < N0 + N1) {
            int q = e - N0;
            int n = q / 512, k = q % 512;
            float f = (k < 256) ? W1[k * 1024 + n] : U1[(k - 256) * 1024 + n];
            wt1[n * 512 + k] = f2bf(f);
        } else {
            int q = e - N0 - N1;
            int n = q / 256, k = q % 256;
            wfcT[n * 256 + k] = f2bf(Wfc[k * 128 + n]);
        }
    }
    if (i < 2) cnt[i] = 0;
}

// ---------------------------------------------------------------------------
// Persistent LSTM layer kernel.
// Grid: 16 WGs (one cluster), 256 threads = 8 wave32 waves each.
// WG `ht` owns H-columns [16*ht, 16*ht+16). Wave v: m-tile = v>>2, gate = v&3.
//
// LDS A-staging is pre-swizzled into WMMA fragment order:
//   hAf[kk][m][lane][slot 0..7 (+4 pad)]  (dwords; 48B lane stride, 16B aligned)
// so the GEMM inner loop is just 2x ds_load_b128 + 1x v_wmma per K-chunk.
// Per-wave weight slice (16 cols x KTOT) is register-resident for all steps.
// 4 independent accumulator chains to hide WMMA latency.
// ---------------------------------------------------------------------------
template <int KIN, bool IN_F32, bool WRITE_SEQ>
__global__ __launch_bounds__(256, 1) void lstm_layer(
    const float* __restrict__ in_f32,      // [B][T][KIN] fp32 (layer 0)
    const unsigned* __restrict__ in_bf16,  // [T][B][KIN]/2 u32 bf16 pairs (layer 1)
    const unsigned* __restrict__ wt,       // [1024][KTOT]/2 u32 bf16 pairs
    const float* __restrict__ bias,        // [1024]
    unsigned* __restrict__ hbuf,           // [2][B][256]/2 u32 bf16 pairs (ping-pong)
    unsigned* __restrict__ seq_out,        // [T][B][256]/2 u32 (if WRITE_SEQ)
    unsigned* __restrict__ cnt) {
    constexpr int KTOT = KIN + 256;
    constexpr int NK = KTOT / 32;
    constexpr int FRAG_U32 = 32 * 12;          // per (kk,m) fragment block (padded)
    constexpr int HAF_U32 = NK * 2 * FRAG_U32;

    extern __shared__ char smem[];
    unsigned* hAf = (unsigned*)smem;                  // swizzled A fragments
    float* zS = (float*)(smem + HAF_U32 * 4);         // [8][16][16] gate pre-acts
    float* cS = zS + 8 * 256;                         // [32][16] cell state

    const int tid = threadIdx.x;
    const int wv = tid >> 5;
    const int lane = tid & 31;
    const int half = lane >> 4;
    const int l16 = lane & 15;
    const int ht = blockIdx.x;   // H tile 0..15
    const int hc0 = ht * 16;
    const int nWG = gridDim.x;

    const int m = wv >> 2;   // M tile (batch rows 16m..16m+15)
    const int g = wv & 3;    // gate: 0=i 1=f 2=g 3=o

    // --- register-resident weight slice (persists all 2048 steps) ---
    Frag bfr[NK];
    {
        const int n = g * 256 + hc0 + l16;  // z column owned by this lane
        const uint4* wq = (const uint4*)wt + (size_t)n * (KTOT / 8) + half * 2;
        #pragma unroll
        for (int kk = 0; kk < NK; ++kk) {
            bfr[kk].q[0] = wq[kk * 4 + 0];
            bfr[kk].q[1] = wq[kk * 4 + 1];
        }
    }
    const float bb = bias[g * 256 + hc0 + l16];

    // --- zero cell state + own slice of hbuf[1] (h_{-1} = 0) ---
    for (int e = tid; e < 512; e += 256) cS[e] = 0.f;
    {
        int b = tid >> 3, n0 = (tid & 7) * 2;
        __hip_atomic_store(&hbuf[(256 * 32 + b * 256 + hc0 + n0) >> 1], 0u,
                           __ATOMIC_RELAXED, __HIP_MEMORY_SCOPE_AGENT);
    }
    unsigned target = 0;
    grid_sync(cnt, target, nWG);

    for (int t = 0; t < 2048; ++t) {
        const int p = t & 1;

        // ---- stage [x_t | h_{t-1}] into LDS, pre-swizzled to fragment order.
        // One K-quad (4 bf16) per iteration -> one ds_store_b64.
        for (int e = tid; e < 32 * KTOT / 4; e += 256) {
            int b = e / (KTOT / 4);
            int k = (e % (KTOT / 4)) * 4;
            unsigned lo, hi;
            if (k < KIN) {
                if (IN_F32) {
                    const float4 s =
                        *(const float4*)(in_f32 + ((size_t)b * 2048 + t) * KIN + k);
                    lo = pack_bf(s.x, s.y);
                    hi = pack_bf(s.z, s.w);
                } else {
                    const uint2 s = *(const uint2*)(in_bf16 +
                                                    ((((size_t)t * 32 + b) * KIN + k) >> 1));
                    lo = s.x;
                    hi = s.y;
                }
            } else {
                int hk = k - KIN;
                const unsigned long long* hp64 =
                    (const unsigned long long*)(hbuf + (1 - p) * 4096 + ((b * 256 + hk) >> 1));
                unsigned long long hv = __hip_atomic_load(hp64, __ATOMIC_RELAXED,
                                                          __HIP_MEMORY_SCOPE_AGENT);
                lo = (unsigned)hv;
                hi = (unsigned)(hv >> 32);
            }
            // (b,k) -> fragment slot: A layout K pair index pr = (k&31)/2
            int mq = b >> 4, l16q = b & 15;
            int kk = k >> 5, pr = (k & 31) >> 1;
            int halfq = (pr >> 3) & 1;
            int vq = pr & 7;   // even (quad = slots v, v+1)
            int laneq = halfq * 16 + l16q;
            int di = (kk * 2 + mq) * FRAG_U32 + laneq * 12 + vq;
            uint2 val;
            val.x = lo;
            val.y = hi;
            *(uint2*)(hAf + di) = val;
        }
        __syncthreads();

        // ---- GEMM: one 16x16 z-tile per wave; 2x b128 LDS load + 1 WMMA per chunk
        //      4 independent accumulator chains (depth NK/4) to hide WMMA latency
        v8f acc[4];
        #pragma unroll
        for (int r = 0; r < 8; ++r) {
            acc[0][r] = bb;
            acc[1][r] = 0.f;
            acc[2][r] = 0.f;
            acc[3][r] = 0.f;
        }
        const uint4* ha4 = (const uint4*)hAf;
        const int abase = m * (FRAG_U32 / 4) + lane * 3;
        #pragma unroll
        for (int kk = 0; kk < NK; ++kk) {
            Frag a;
            a.q[0] = ha4[kk * (FRAG_U32 / 2) + abase + 0];
            a.q[1] = ha4[kk * (FRAG_U32 / 2) + abase + 1];
            acc[kk & 3] = __builtin_amdgcn_wmma_f32_16x16x32_bf16(
                false, a.v, false, bfr[kk].v, (short)0, acc[kk & 3], false, false);
        }
        #pragma unroll
        for (int r = 0; r < 8; ++r)
            zS[(m * 4 + g) * 256 + (r + 8 * half) * 16 + l16] =
                (acc[0][r] + acc[1][r]) + (acc[2][r] + acc[3][r]);
        __syncthreads();

        // ---- elementwise LSTM update (linear act, sigmoid recurrent act) ----
        {
            int b = tid >> 3, n0 = (tid & 7) * 2;
            int mm = b >> 4, zr = b & 15;
            float h2[2];
            #pragma unroll
            for (int j = 0; j < 2; ++j) {
                int n = n0 + j;
                int zb = zr * 16 + n;
                float zi = zS[(mm * 4 + 0) * 256 + zb];
                float zf = zS[(mm * 4 + 1) * 256 + zb];
                float zg = zS[(mm * 4 + 2) * 256 + zb];
                float zo = zS[(mm * 4 + 3) * 256 + zb];
                float c0 = cS[b * 16 + n];
                float cn = sigf(zf) * c0 + sigf(zi) * zg;  // candidate act = linear
                cS[b * 16 + n] = cn;
                h2[j] = sigf(zo) * cn;                     // output act = linear
            }
            unsigned hp32 = pack_bf(h2[0], h2[1]);
            __hip_atomic_store(&hbuf[(p * (256 * 32) + b * 256 + hc0 + n0) >> 1], hp32,
                               __ATOMIC_RELAXED, __HIP_MEMORY_SCOPE_AGENT);
            if (WRITE_SEQ)
                seq_out[(((size_t)t * 32 + b) * 256 + hc0 + n0) >> 1] = hp32;
        }
        grid_sync(cnt, target, nWG);
    }
}

// ---------------------------------------------------------------------------
// FC: out[32,128] = hT[32,256] @ Wfc[256,128] + bfc  (bf16 WMMA, one WG)
// ---------------------------------------------------------------------------
__global__ __launch_bounds__(256) void fc_kernel(const unsigned* __restrict__ hT,
                                                 const unsigned* __restrict__ wfcT,
                                                 const float* __restrict__ bfc,
                                                 float* __restrict__ out) {
    __shared__ unsigned hA[32 * 128];  // 32x256 bf16
    const int tid = threadIdx.x;
    const int wv = tid >> 5, lane = tid & 31, half = lane >> 4, l16 = lane & 15;
    for (int e = tid; e < 32 * 128; e += 256) hA[e] = hT[e];
    __syncthreads();

    const int nt = wv;  // out-col tile 0..7
    const float bb = bfc[nt * 16 + l16];
    v8f acc0, acc1;
    #pragma unroll
    for (int r = 0; r < 8; ++r) { acc0[r] = bb; acc1[r] = bb; }
    const uint4* wq = (const uint4*)wfcT + (size_t)(nt * 16 + l16) * 32 + half * 2;
    #pragma unroll
    for (int kk = 0; kk < 8; ++kk) {
        Frag bf, a0, a1;
        bf.q[0] = wq[kk * 4 + 0];
        bf.q[1] = wq[kk * 4 + 1];
        #pragma unroll
        for (int v = 0; v < 8; ++v) {
            int k0 = kk * 32 + 16 * half + 2 * v;
            a0.u[v] = hA[(l16 * 256 + k0) >> 1];
            a1.u[v] = hA[((16 + l16) * 256 + k0) >> 1];
        }
        acc0 = __builtin_amdgcn_wmma_f32_16x16x32_bf16(false, a0.v, false, bf.v,
                                                       (short)0, acc0, false, false);
        acc1 = __builtin_amdgcn_wmma_f32_16x16x32_bf16(false, a1.v, false, bf.v,
                                                       (short)0, acc1, false, false);
    }
    #pragma unroll
    for (int r = 0; r < 8; ++r) {
        out[(r + 8 * half) * 128 + nt * 16 + l16] = acc0[r];
        out[(16 + r + 8 * half) * 128 + nt * 16 + l16] = acc1[r];
    }
}

// ---------------------------------------------------------------------------
// Host launcher
// ---------------------------------------------------------------------------
extern "C" void kernel_launch(void* const* d_in, const int* in_sizes, int n_in,
                              void* d_out, int out_size, void* d_ws, size_t ws_size,
                              hipStream_t stream) {
    (void)in_sizes; (void)n_in; (void)out_size; (void)ws_size;
    const float* x   = (const float*)d_in[0];
    const float* W0  = (const float*)d_in[1];
    const float* U0  = (const float*)d_in[2];
    const float* b0  = (const float*)d_in[3];
    const float* W1  = (const float*)d_in[4];
    const float* U1  = (const float*)d_in[5];
    const float* b1  = (const float*)d_in[6];
    const float* Wfc = (const float*)d_in[7];
    const float* bfc = (const float*)d_in[8];
    float* out = (float*)d_out;

    char* ws = (char*)d_ws;
    size_t o = 0;
    unsigned short* wt0  = (unsigned short*)(ws + o); o += (size_t)1024 * 384 * 2;
    unsigned short* wt1  = (unsigned short*)(ws + o); o += (size_t)1024 * 512 * 2;
    unsigned short* wfcT = (unsigned short*)(ws + o); o += (size_t)128 * 256 * 2;
    unsigned* hbuf = (unsigned*)(ws + o);             o += (size_t)2 * 32 * 256 * 2;
    unsigned* cnt  = (unsigned*)(ws + o);             o += 256;
    unsigned* h0seq = (unsigned*)(ws + o);            o += (size_t)2048 * 32 * 256 * 2;

    convert_kernel<<<512, 256, 0, stream>>>(W0, U0, W1, U1, Wfc, wt0, wt1, wfcT, cnt);

    const size_t sm0 = (size_t)12 * 2 * 32 * 12 * 4 + 8 * 256 * 4 + 512 * 4;  // 47104 B
    lstm_layer<128, true, true><<<16, 256, sm0, stream>>>(
        x, nullptr, (const unsigned*)wt0, b0, hbuf, h0seq, cnt + 0);

    const size_t sm1 = (size_t)16 * 2 * 32 * 12 * 4 + 8 * 256 * 4 + 512 * 4;  // 59392 B
    lstm_layer<256, false, false><<<16, 256, sm1, stream>>>(
        nullptr, h0seq, (const unsigned*)wt1, b1, hbuf, nullptr, cnt + 1);

    fc_kernel<<<1, 256, 0, stream>>>(hbuf + 4096, (const unsigned*)wfcT, bfc, out);
}
